// StructuredAuxDeployModule_2293512536127
// MI455X (gfx1250) — compile-verified, matrix operand order checked by reading, not run on hardware
//
#include <hip/hip_runtime.h>

// ---------------------------------------------------------------------------
// StructuredAuxDeployModule on gfx1250: 4 small MLPs (7/7/6/5 -> 64 -> 64 ->
// 1/1/1/3) over B=1M rows. ALL three layers run on v_wmma_f32_16x16x32_f16:
//   L1: K zero-padded 7->32 (zeros from LDS zero pad -> direct tuple loads)
//   L2: true 64x64, 8 WMMAs
//   L3: N zero-padded (1|3)->16 (zero columns baked into W3t)
// All WMMA chains start from C = 0 (inline-0 SRC2); biases are folded into
// the activation epilogue (pre-scaled by 2*log2e) or added at read-out (L3).
// Intra-wave LDS hand-offs use wave_barrier (no s_barrier): per-tile buffers
// are wave-private and CDNA5 DS ops from one wave execute in order.
// ---------------------------------------------------------------------------

typedef _Float16 v8h  __attribute__((ext_vector_type(8)));
typedef _Float16 v16h __attribute__((ext_vector_type(16)));
typedef float    v8f  __attribute__((ext_vector_type(8)));

#define NW            4            // waves per block (wave32)
#define BLOCK         (NW * 32)    // 128 threads
#define ROWS_PER_TILE (NW * 16)    // 64 rows per block-iteration

#define TANH_K 2.8853900817779268f // 2*log2(e)

struct Params {
    const float* p[30];
    float* out_ut;     // (B,2)
    float* out_fr;     // (B,1)
    float* out_res;    // (B,3)
    int nrows;
    int tiles;
};

struct Smem {
    _Float16 W1t[4][64][16] __attribute__((aligned(16)));  // [mlp][n][k], k>=kin zeroed   (8 KB)
    float    B1s[4][64];                                   // bias * 2log2e                (1 KB)
    _Float16 W2t[4][64][64] __attribute__((aligned(16)));  // [mlp][n][k]                  (32 KB)
    float    B2s[4][64];                                   // bias * 2log2e                (1 KB)
    _Float16 W3t[4][16][64] __attribute__((aligned(16)));  // [mlp][o(pad16)][k], o>=OD 0  (8 KB)
    float    B3p[4][16];                                   // bias padded with zeros       (256 B)
    float    mean[5], stdv[5], inv[5], tmean[3], tstd[3];
    _Float16 zpad[16]       __attribute__((aligned(16)));  // 32 B zeros: frag upper halves
    _Float16 featH[NW][16][16] __attribute__((aligned(16)));// per-wave feats f16, k pad 16 (2 KB)
    _Float16 H[NW][16][64]   __attribute__((aligned(16))); // per-wave activations         (8 KB)
    float    outb[NW][16][4];                              // layer-3 results              (1 KB)
};

// tanh(acc + b) with pre-scaled bias b2 = b * 2log2e:
// tanh(x) = 1 - 2/(exp2(2x*log2e)+1); exact limits at +-inf, NaN-propagating.
__device__ __forceinline__ float fast_tanh_pre(float acc, float b2) {
    float e = __builtin_amdgcn_exp2f(fmaf(acc, TANH_K, b2));
    return 1.0f - 2.0f * __builtin_amdgcn_rcpf(e + 1.0f);
}

__device__ __forceinline__ float fast_tanh(float x) {
    float e = __builtin_amdgcn_exp2f(x * TANH_K);
    return 1.0f - 2.0f * __builtin_amdgcn_rcpf(e + 1.0f);
}

__device__ __forceinline__ float softplusf(float x) {
    float e = __builtin_amdgcn_exp2f(x * 1.4426950408889634f);
    float r = 0.6931471805599453f * __builtin_amdgcn_logf(1.0f + e);
    return (x > 20.0f) ? x : r;
}

// Compiler-only fence: per-tile LDS buffers are wave-private and the DS unit
// processes one wave's LDS ops in order; we only must stop code motion.
__device__ __forceinline__ void wave_fence() { __builtin_amdgcn_wave_barrier(); }

// 16-half WMMA fragment from two 16B LDS reads. Per the CDNA5 16-bit A/B
// layout, VGPR0..3 hold K=kb..kb+7 (pairs), VGPR4..7 hold K=kb+16..kb+23 ->
// simple concat of two 8-half runs; both halves come from LDS so the register
// allocator can place the ds_load_b128 results directly into the operand tuple.
__device__ __forceinline__ v16h ld_frag2(const _Float16* lop, const _Float16* hip_) {
    v8h lo = *(const v8h*)lop;
    v8h hi = *(const v8h*)hip_;
    v16h r;
#pragma unroll
    for (int i = 0; i < 8; i++) { r[i] = lo[i]; r[i + 8] = hi[i]; }
    return r;
}
__device__ __forceinline__ v16h ld_frag(const _Float16* rowp, int kb) {
    return ld_frag2(rowp + kb, rowp + kb + 16);
}

template <int OD>
__device__ __forceinline__ void mlp_run(Smem& sm, int m, int wid, int lane, float* out) {
    const int row16  = lane & 15;
    const int hiHalf = lane >> 4;   // 0 or 1
    const int kb     = hiHalf << 3; // +0 (lanes 0-15) / +8 (lanes 16-31)
    const int mbase  = hiHalf << 3; // D-fragment row base
    const v8f  zc    = {};          // zero accumulator (inline-0 SRC2)

    // ---- Layer 1: (16 x 7pad32) @ (7pad32 x 64) : 4 WMMAs, tanh -> H ----
    const v16h a = ld_frag2(&sm.featH[wid][row16][kb], &sm.zpad[0]);
#pragma unroll
    for (int nt = 0; nt < 4; nt++) {
        const int n = nt * 16 + row16;
        const v16h b = ld_frag2(&sm.W1t[m][n][kb], &sm.zpad[0]);
        const v8f c = __builtin_amdgcn_wmma_f32_16x16x32_f16(false, a, false, b, (short)0, zc, false, false);
        const float b2 = sm.B1s[m][n];
#pragma unroll
        for (int v = 0; v < 8; v++)
            sm.H[wid][mbase + v][n] = (_Float16)fast_tanh_pre(c[v], b2);
    }
    wave_fence();

    // ---- Layer 2: (16x64)@(64x64) : 8 WMMAs, tanh -> H (in place) ----
    const v16h a0 = ld_frag(&sm.H[wid][row16][0], kb);       // K-chunk 0
    const v16h a1 = ld_frag(&sm.H[wid][row16][0], kb + 32);  // K-chunk 1
    wave_fence();  // A fragments in regs before H is overwritten
#pragma unroll
    for (int nt = 0; nt < 4; nt++) {
        const int n = nt * 16 + row16;
        const v16h b0 = ld_frag(&sm.W2t[m][n][0], kb);
        const v16h b1 = ld_frag(&sm.W2t[m][n][0], kb + 32);
        v8f c = __builtin_amdgcn_wmma_f32_16x16x32_f16(false, a0, false, b0, (short)0, zc, false, false);
        c     = __builtin_amdgcn_wmma_f32_16x16x32_f16(false, a1, false, b1, (short)0, c,  false, false);
        const float b2 = sm.B2s[m][n];
#pragma unroll
        for (int v = 0; v < 8; v++)
            sm.H[wid][mbase + v][n] = (_Float16)fast_tanh_pre(c[v], b2);
    }
    wave_fence();

    // ---- Layer 3: (16x64)@(64 x ODpad16) : 2 WMMAs (linear) ----
    const v16h h0 = ld_frag(&sm.H[wid][row16][0], kb);
    const v16h h1 = ld_frag(&sm.H[wid][row16][0], kb + 32);
    const v16h w0 = ld_frag(&sm.W3t[m][row16][0], kb);
    const v16h w1 = ld_frag(&sm.W3t[m][row16][0], kb + 32);
    v8f c = __builtin_amdgcn_wmma_f32_16x16x32_f16(false, h0, false, w0, (short)0, zc, false, false);
    c     = __builtin_amdgcn_wmma_f32_16x16x32_f16(false, h1, false, w1, (short)0, c,  false, false);
    if (row16 < OD) {   // lane holds output column N=row16; VGPR v holds row M=mbase+v
#pragma unroll
        for (int v = 0; v < 8; v++)
            sm.outb[wid][mbase + v][row16] = c[v];
    }
    wave_fence();
    if (lane < 16) {
#pragma unroll
        for (int o = 0; o < OD; o++) out[o] = sm.outb[wid][lane][o] + sm.B3p[m][o];
    }
}

// lanes 0-15: write this row's features (f16; slots 7..15 stay zero — slot 7
// is written 0 here, slots 8..15 are zero-initialized once at kernel start).
__device__ __forceinline__ void write_feat(Smem& sm, int wid, int lane,
                                           float f0, float f1, float f2, float f3,
                                           float f4, float f5, float f6) {
    if (lane < 16) {
        v8h v;
        v[0] = (_Float16)f0; v[1] = (_Float16)f1; v[2] = (_Float16)f2; v[3] = (_Float16)f3;
        v[4] = (_Float16)f4; v[5] = (_Float16)f5; v[6] = (_Float16)f6; v[7] = (_Float16)0.f;
        *(v8h*)&sm.featH[wid][lane][0] = v;
    }
}

__global__ void __launch_bounds__(BLOCK) aux_deploy_kernel(Params P) {
    __shared__ Smem sm;
    const int tid  = threadIdx.x;
    const int wid  = tid >> 5;
    const int lane = tid & 31;

    // ---- Stage all weights into LDS once per block (hits L2/WGP$) ----
    if (tid < 5) { sm.mean[tid] = P.p[2][tid]; float s = P.p[3][tid]; sm.stdv[tid] = s; sm.inv[tid] = 1.0f / s; }
    else if (tid >= 8 && tid < 11) { int i = tid - 8; sm.tmean[i] = P.p[4][i]; sm.tstd[i] = P.p[5][i]; }
    if (tid < 16) sm.zpad[tid] = (_Float16)0.f;
#pragma unroll
    for (int m = 0; m < 4; m++) {
        const int KINS[4] = {7, 7, 6, 5};
        const int ODS[4]  = {1, 1, 1, 3};
        const float* W1 = P.p[6 + 6 * m];
        const float* b1 = P.p[7 + 6 * m];
        const float* W2 = P.p[8 + 6 * m];
        const float* b2 = P.p[9 + 6 * m];
        const float* W3 = P.p[10 + 6 * m];
        const float* b3 = P.p[11 + 6 * m];
        const int kin = KINS[m], od = ODS[m];
        for (int i = tid; i < 64 * 16; i += BLOCK) {           // W1 transpose + K zero-pad
            const int n = i >> 4, k = i & 15;
            sm.W1t[m][n][k] = (k < kin) ? (_Float16)W1[k * 64 + n] : (_Float16)0.f;
        }
        if (tid < 64) { sm.B1s[m][tid] = b1[tid] * TANH_K; sm.B2s[m][tid] = b2[tid] * TANH_K; }
        for (int i = tid; i < 4096; i += BLOCK)                // W2 transpose [k][n]->[n][k]
            sm.W2t[m][i & 63][i >> 6] = (_Float16)W2[i];
        for (int i = tid; i < 16 * 64; i += BLOCK) {           // W3 transpose + N zero-pad
            const int o = i >> 6, k = i & 63;
            sm.W3t[m][o][k] = (o < od) ? (_Float16)W3[k * od + o] : (_Float16)0.f;
        }
        if (tid < 16) sm.B3p[m][tid] = (tid < od) ? b3[tid] : 0.f;
    }
    // featH K-slots 8..15 are permanently zero (read by lanes 16-31)
    if (lane < 16) { v8h z = {}; *(v8h*)&sm.featH[wid][lane][8] = z; }
    __syncthreads();   // only cross-wave sync in the kernel

    for (int t = blockIdx.x; t < P.tiles; t += gridDim.x) {
        const long long base = (long long)t * ROWS_PER_TILE + wid * 16;
        float vx = 0, vy = 0, w = 0, vel = 0, delta = 0, smag = 0, sgn = 0;
        long long r = 0; bool valid = false;

        if (lane < 16) {
            r = base + lane;
            valid = r < (long long)P.nrows;
            const long long rr = valid ? r : 0;
            const float* xd = P.p[0] + rr * 3;
            const float* ut = P.p[1] + rr * 2;
            vx    = (xd[0] - sm.mean[0]) * sm.inv[0];
            vy    = (xd[1] - sm.mean[1]) * sm.inv[1];
            w     = (xd[2] - sm.mean[2]) * sm.inv[2];
            vel   = (ut[0] - sm.mean[3]) * sm.inv[3];
            delta = (ut[1] - sm.mean[4]) * sm.inv[4];
            smag  = __builtin_amdgcn_sqrtf(vx * vx + vy * vy + 1e-8f);
            sgn   = (vx > 0.f) ? 1.f : ((vx < 0.f) ? -1.f : 0.f);
        }
        write_feat(sm, wid, lane, delta, vel, vx, vy, w, smag, sgn);      // steer_in
        wave_fence();

        float o1[3] = {0.f, 0.f, 0.f};
        float d_steer = 0.f;
        mlp_run<1>(sm, 0, wid, lane, o1);                                  // steer MLP
        if (lane < 16) d_steer = 0.5f * fast_tanh(o1[0]);
        write_feat(sm, wid, lane, vel, delta, vx, smag, sgn, vy, w);      // acc_in
        wave_fence();

        mlp_run<1>(sm, 1, wid, lane, o1);                                  // acc MLP
        float ute0 = 0.f, ute1 = 0.f;
        if (lane < 16) {
            const float d_acc = 0.5f * fast_tanh(o1[0]);
            ute0 = vel + d_acc;
            ute1 = delta + d_steer;
        }
        write_feat(sm, wid, lane, vx, vy, w, ute0, ute1, 0.02f, 0.f);     // fric_in
        wave_fence();

        mlp_run<1>(sm, 2, wid, lane, o1);                                  // friction MLP
        float fric = 0.f;
        if (lane < 16) fric = 1.0f + softplusf(o1[0]);
        write_feat(sm, wid, lane, vx, vy, w, ute0, ute1, 0.f, 0.f);       // res_in
        wave_fence();

        float o3[3] = {0.f, 0.f, 0.f};
        mlp_run<3>(sm, 3, wid, lane, o3);                                  // residual MLP

        if (lane < 16 && valid) {
            P.out_ut[r * 2 + 0] = ute0 * sm.stdv[3] + sm.mean[3];
            P.out_ut[r * 2 + 1] = ute1 * sm.stdv[4] + sm.mean[4];
            P.out_fr[r] = fric;
#pragma unroll
            for (int o = 0; o < 3; o++) P.out_res[r * 3 + o] = o3[o] * sm.tstd[o] + sm.tmean[o];
        }
        wave_fence();
    }
}

extern "C" void kernel_launch(void* const* d_in, const int* in_sizes, int n_in,
                              void* d_out, int out_size, void* d_ws, size_t ws_size,
                              hipStream_t stream) {
    (void)n_in; (void)d_ws; (void)ws_size; (void)out_size;
    Params P;
    for (int i = 0; i < 30; i++) P.p[i] = (const float*)d_in[i];
    const int nrows = in_sizes[0] / 3;
    float* out = (float*)d_out;
    P.out_ut  = out;
    P.out_fr  = out + (size_t)nrows * 2;
    P.out_res = out + (size_t)nrows * 3;
    P.nrows   = nrows;
    P.tiles   = (nrows + ROWS_PER_TILE - 1) / ROWS_PER_TILE;
    int grid = P.tiles < 2048 ? P.tiles : 2048;
    if (grid < 1) grid = 1;
    aux_deploy_kernel<<<grid, BLOCK, 0, stream>>>(P);
}
